// MoE_16879221473729
// MI455X (gfx1250) — matrix operand order repaired and checked by reading, not compiled, not run on hardware
//
#include <hip/hip_runtime.h>
#include <hip/hip_bf16.h>

// MoE top-2 SwiGLU for MI455X (gfx1250).
// Strategy: route tokens (top-2 of 8 experts), convert x + all expert weights to
// bf16 once per call (weights drop 201MB->100MB, L2-resident at 192MB), then run
// the three GEMMs per token-tile with v_wmma_f32_16x16x32_bf16, fp32 accumulate.
// Compute-bound: ~206 GFLOP routed vs ~9us of HBM traffic at 23.3 TB/s.

typedef __attribute__((ext_vector_type(16))) __bf16       v16bf;
typedef __attribute__((ext_vector_type(8)))  float        v8f;
typedef __attribute__((ext_vector_type(8)))  unsigned int v8u;
typedef __attribute__((ext_vector_type(4)))  int          v4i_;

#define N_TOK 8192
#define HDIM  1024
#define IDIM  2048
#define EXP   8

__device__ __forceinline__ unsigned short f32_to_bf16(float f) {
    unsigned u = __float_as_uint(f);
    unsigned r = u + 0x7FFFu + ((u >> 16) & 1u);   // round-to-nearest-even
    return (unsigned short)(r >> 16);
}

// silu with hardware v_rcp_f32 (activation precision is fine; avoids the
// v_div_scale/Newton-iteration chain the IEEE divide lowers to)
__device__ __forceinline__ float fast_silu(float g) {
    return g * __builtin_amdgcn_rcpf(1.f + __expf(-g));
}

// ---------------- gfx1250 async global->LDS copy (ASYNCcnt-tracked) ---------
__device__ __forceinline__ void async_copy16(const uint4* gsrc, uint4* ldst) {
#if __has_builtin(__builtin_amdgcn_global_load_async_to_lds_b128)
    __builtin_amdgcn_global_load_async_to_lds_b128(
        (__attribute__((address_space(1))) v4i_*)(unsigned long)gsrc,
        (__attribute__((address_space(3))) v4i_*)ldst, 0, 0);
#else
    unsigned lds_off = (unsigned)(unsigned long)(__attribute__((address_space(3))) v4i_*)ldst;
    asm volatile("global_load_async_to_lds_b128 %0, %1, off"
                 :: "v"(lds_off), "v"((unsigned long)gsrc) : "memory");
#endif
}
__device__ __forceinline__ void async_wait0() {
#if __has_builtin(__builtin_amdgcn_s_wait_asynccnt)
    __builtin_amdgcn_s_wait_asynccnt(0);
#else
    asm volatile("s_wait_asynccnt 0x0" ::: "memory");
#endif
}

// ---------------------------------------------------------------- zero ------
__global__ void zero_f(float* p, long n) {
    long i = (long)blockIdx.x * blockDim.x + threadIdx.x;
    long stride = (long)gridDim.x * blockDim.x;
    for (; i < n; i += stride) p[i] = 0.f;
}

// ------------------------------------------------------- x -> bf16 ----------
__global__ void cvt_x_kernel(const float* __restrict__ x, unsigned short* __restrict__ xb, long n) {
    long i = (long)blockIdx.x * blockDim.x + threadIdx.x;
    long stride = (long)gridDim.x * blockDim.x;
    for (; i < n; i += stride) xb[i] = f32_to_bf16(x[i]);
}

// ------------------------------ per-expert transpose + fp32->bf16 -----------
// src: [gridDim.z][R][C] fp32  ->  dst: [gridDim.z][C][R] bf16
__global__ __launch_bounds__(256) void transpose_cvt_kernel(
    const float* __restrict__ src, unsigned short* __restrict__ dst, int R, int C) {
    __shared__ float tile[32][33];
    const float* s = src + (long)blockIdx.z * R * C;
    unsigned short* d = dst + (long)blockIdx.z * R * C;
    int c0 = blockIdx.x * 32, r0 = blockIdx.y * 32;
    for (int i = threadIdx.y; i < 32; i += 8)
        tile[i][threadIdx.x] = s[(long)(r0 + i) * C + c0 + threadIdx.x];
    __syncthreads();
    for (int i = threadIdx.y; i < 32; i += 8)
        d[(long)(c0 + i) * R + r0 + threadIdx.x] = f32_to_bf16(tile[threadIdx.x][i]);
}

// ---------------------------------------------------------------- router ----
// One wave per token: 8 dot-products over H=1024, softmax fp32, top-2,
// renormalize, scatter into per-expert compacted token lists.
__global__ __launch_bounds__(256) void router_kernel(
    const float* __restrict__ x, const float* __restrict__ gw,
    int* cnt, float* loadv, float* countsv, int* etok, float* ewt) {
    int wave = threadIdx.x >> 5, lid = threadIdx.x & 31;
    int t = blockIdx.x * 8 + wave;
    const float* xr = x + (long)t * HDIM;
    float acc[EXP];
#pragma unroll
    for (int e = 0; e < EXP; ++e) acc[e] = 0.f;
    for (int h = lid; h < HDIM; h += 32) {
        float xv = xr[h];
        const float* g = gw + h * EXP;
#pragma unroll
        for (int e = 0; e < EXP; ++e) acc[e] += xv * g[e];
    }
#pragma unroll
    for (int e = 0; e < EXP; ++e)
        for (int off = 16; off; off >>= 1) acc[e] += __shfl_xor(acc[e], off, 32);
    if (lid == 0) {
        float mx = acc[0];
#pragma unroll
        for (int e = 1; e < EXP; ++e) mx = fmaxf(mx, acc[e]);
        float p[EXP], s = 0.f;
#pragma unroll
        for (int e = 0; e < EXP; ++e) { p[e] = __expf(acc[e] - mx); s += p[e]; }
        float inv = 1.f / s;
#pragma unroll
        for (int e = 0; e < EXP; ++e) p[e] *= inv;
        int i0 = 0;
#pragma unroll
        for (int e = 1; e < EXP; ++e) if (p[e] > p[i0]) i0 = e;
        int i1 = (i0 == 0) ? 1 : 0;
#pragma unroll
        for (int e = 0; e < EXP; ++e) if (e != i0 && p[e] > p[i1]) i1 = e;
        float w0 = p[i0], w1 = p[i1];
        float rs = 1.f / (w0 + w1);
        // aux-loss accumulators use pre-renormalization router weights
        atomicAdd(&loadv[i0], w0); atomicAdd(&loadv[i1], w1);
        atomicAdd(&countsv[i0], 1.f); atomicAdd(&countsv[i1], 1.f);
        int p0 = atomicAdd(&cnt[i0], 1); etok[i0 * N_TOK + p0] = t; ewt[i0 * N_TOK + p0] = w0 * rs;
        int p1 = atomicAdd(&cnt[i1], 1); etok[i1 * N_TOK + p1] = t; ewt[i1 * N_TOK + p1] = w1 * rs;
    }
}

// ---------------------------------------------------------------- aux -------
__global__ void aux_kernel(const float* loadv, const float* countsv, float* out_aux) {
    if (threadIdx.x == 0 && blockIdx.x == 0) {
        float s = 0.f;
        for (int e = 0; e < EXP; ++e)
            s += (loadv[e] / (float)N_TOK) * (countsv[e] / (float)(N_TOK * 2));
        out_aux[0] = s * (float)EXP * 1e-3f;
    }
}

// ---------------------------------------------------------------- FFN -------
// Block = 16 tokens of one expert, 8 waves, full H output.
// Per I-chunk of 256: waves cooperatively build Hc = silu(X*Wg) ⊙ (X*Wu) in LDS
// (each wave: 2 tiles of 16 I-cols, K=1024), then each wave accumulates its
// 128-column slice of Out += Hc * Wd (K=256 per chunk).
__global__ __launch_bounds__(256) void moe_ffn_kernel(
    const unsigned short* __restrict__ xb,
    const unsigned short* __restrict__ wgt,   // [E][I][H] bf16
    const unsigned short* __restrict__ wut,   // [E][I][H] bf16
    const unsigned short* __restrict__ wdt,   // [E][H][I] bf16
    const int* __restrict__ cnt, const int* __restrict__ etok,
    const float* __restrict__ ewt, float* __restrict__ y) {
    const int tilesPer = N_TOK / 16;                       // 512
    int e   = blockIdx.x / tilesPer;
    int tix = blockIdx.x % tilesPer;
    int c   = cnt[e];
    if (tix * 16 >= c) return;                             // block-uniform exit

    __shared__ int   sTok[16];
    __shared__ float sWt[16];
    __shared__ __align__(16) unsigned short sX[16 * HDIM]; // 32 KB
    __shared__ __align__(16) unsigned short sHc[16 * 256]; //  8 KB

    int tid = threadIdx.x;
    if (tid < 16) {
        int r = tix * 16 + tid;
        int rr = (r < c) ? r : (c - 1);
        sTok[tid] = etok[e * N_TOK + rr];
        sWt[tid]  = (r < c) ? ewt[e * N_TOK + rr] : 0.f;
    }
    __syncthreads();
    {   // gather X tile rows into LDS via async DMA (ASYNCcnt path)
        const uint4* xs = (const uint4*)xb;                // row = 128 uint4
        uint4* xd = (uint4*)sX;
        for (int i = tid; i < 16 * 128; i += 256) {
            int r = i >> 7, ci = i & 127;
            async_copy16(xs + (long)sTok[r] * 128 + ci, xd + i);
        }
        async_wait0();
    }
    __syncthreads();

    int lid = tid & 31, wave = tid >> 5;
    int m = lid & 15, hi = lid >> 4;                       // WMMA lane coords
    const unsigned int* sXu = (const unsigned int*)sX;     // [16][512] u32
    const unsigned int* sHu = (const unsigned int*)sHc;    // [16][128] u32
    const unsigned int* wgu = ((const unsigned int*)wgt) + (long)e * IDIM * (HDIM / 2);
    const unsigned int* wuu = ((const unsigned int*)wut) + (long)e * IDIM * (HDIM / 2);
    const unsigned int* wdu = ((const unsigned int*)wdt) + (long)e * HDIM * (IDIM / 2);

    v8f accD[8];
#pragma unroll
    for (int nt = 0; nt < 8; ++nt) accD[nt] = (v8f){};

    for (int ch = 0; ch < 8; ++ch) {
        int i0 = ch * 256;
        int n0 = i0 + wave * 32;                           // this wave's I cols
        // warm L2/WGP$ for next chunk's gate/up weight rows
        if (ch < 7) {
            long pf = (long)(n0 + 256 + m) * (HDIM / 2);
            __builtin_prefetch(wgu + pf, 0, 1);
            __builtin_prefetch(wuu + pf, 0, 1);
        }
        v8f g0 = (v8f){}, u0 = (v8f){}, g1 = (v8f){}, u1 = (v8f){};
#pragma unroll 1
        for (int k = 0; k < 32; ++k) {
            int kb = k * 32;
            // A fragment (16x32 bf16) from LDS, contiguous K order (k = 16*hi + s)
            v8u au;
            int ab = m * 512 + (kb >> 1) + hi * 8;
#pragma unroll
            for (int j = 0; j < 4; ++j) { au[j] = sXu[ab + j]; au[4 + j] = sXu[ab + 4 + j]; }
            v16bf av = __builtin_bit_cast(v16bf, au);
            // B fragments (32x16 bf16): row = I col, contiguous K pairs over H
            long rb = (long)(n0 + m) * (HDIM / 2) + (kb >> 1) + hi * 8;
            v8u bg0, bu0, bg1, bu1;
#pragma unroll
            for (int j = 0; j < 8; ++j) {
                bg0[j] = wgu[rb + j];
                bu0[j] = wuu[rb + j];
                bg1[j] = wgu[rb + 16 * (HDIM / 2) + j];
                bu1[j] = wuu[rb + 16 * (HDIM / 2) + j];
            }
            g0 = __builtin_amdgcn_wmma_f32_16x16x32_bf16(false, av, false, __builtin_bit_cast(v16bf, bg0), (short)0, g0, false, false);
            u0 = __builtin_amdgcn_wmma_f32_16x16x32_bf16(false, av, false, __builtin_bit_cast(v16bf, bu0), (short)0, u0, false, false);
            g1 = __builtin_amdgcn_wmma_f32_16x16x32_bf16(false, av, false, __builtin_bit_cast(v16bf, bg1), (short)0, g1, false, false);
            u1 = __builtin_amdgcn_wmma_f32_16x16x32_bf16(false, av, false, __builtin_bit_cast(v16bf, bu1), (short)0, u1, false, false);
        }
        __syncthreads();                                   // prev chunk pass2 done
#pragma unroll
        for (int j = 0; j < 8; ++j) {                      // silu(g)*u -> LDS bf16
            int row = j + 8 * hi;
            sHc[row * 256 + wave * 32 + m]      = f32_to_bf16(fast_silu(g0[j]) * u0[j]);
            sHc[row * 256 + wave * 32 + 16 + m] = f32_to_bf16(fast_silu(g1[j]) * u1[j]);
        }
        __syncthreads();
        // down-projection: this wave's 128 output columns, K = this I chunk
#pragma unroll 1
        for (int k2 = 0; k2 < 8; ++k2) {
            int kk = k2 * 32;
            v8u au;
            int ab = m * 128 + (kk >> 1) + hi * 8;
#pragma unroll
            for (int j = 0; j < 4; ++j) { au[j] = sHu[ab + j]; au[4 + j] = sHu[ab + 4 + j]; }
            v16bf av = __builtin_bit_cast(v16bf, au);
#pragma unroll
            for (int nt = 0; nt < 8; ++nt) {
                int hcol = wave * 128 + nt * 16 + m;
                long rb = (long)hcol * (IDIM / 2) + ((i0 + kk) >> 1) + hi * 8;
                v8u b;
#pragma unroll
                for (int j = 0; j < 8; ++j) b[j] = wdu[rb + j];
                accD[nt] = __builtin_amdgcn_wmma_f32_16x16x32_bf16(false, av, false, __builtin_bit_cast(v16bf, b), (short)0, accD[nt], false, false);
            }
        }
    }
    // epilogue: scale by router weight, accumulate into y (token may hit 2 experts)
#pragma unroll
    for (int nt = 0; nt < 8; ++nt) {
        int col = wave * 128 + nt * 16 + m;
#pragma unroll
        for (int j = 0; j < 8; ++j) {
            int row = j + 8 * hi;
            float v = accD[nt][j] * sWt[row];
            unsafeAtomicAdd(&y[(long)sTok[row] * HDIM + col], v);
        }
    }
}

// ---------------------------------------------------------------- launch ----
extern "C" void kernel_launch(void* const* d_in, const int* in_sizes, int n_in,
                              void* d_out, int out_size, void* d_ws, size_t ws_size,
                              hipStream_t stream) {
    const float* x      = (const float*)d_in[0];
    const float* gate_w = (const float*)d_in[1];
    const float* w_gate = (const float*)d_in[2];
    const float* w_up   = (const float*)d_in[3];
    const float* w_down = (const float*)d_in[4];
    float* y = (float*)d_out;                   // [N][H] then aux scalar

    // workspace carve-up (~113 MB)
    char* w = (char*)d_ws;
    int*   cnt     = (int*)w;            w += 256;
    float* loadv   = (float*)w;          w += 256;
    float* countsv = (float*)w;          w += 256;
    int*   etok    = (int*)w;            w += (long)EXP * N_TOK * 4;
    float* ewt     = (float*)w;          w += (long)EXP * N_TOK * 4;
    unsigned short* xb  = (unsigned short*)w; w += (long)N_TOK * HDIM * 2;
    unsigned short* wgt = (unsigned short*)w; w += (long)EXP * IDIM * HDIM * 2;
    unsigned short* wut = (unsigned short*)w; w += (long)EXP * IDIM * HDIM * 2;
    unsigned short* wdt = (unsigned short*)w; w += (long)EXP * HDIM * IDIM * 2;
    (void)in_sizes; (void)n_in; (void)out_size; (void)ws_size;

    // 1. zero output + counters
    zero_f<<<2048, 256, 0, stream>>>(y, (long)N_TOK * HDIM + 1);
    zero_f<<<1, 256, 0, stream>>>((float*)d_ws, 192);

    // 2. convert activations + weights to bf16 (weights transposed for B frags)
    cvt_x_kernel<<<4096, 256, 0, stream>>>(x, xb, (long)N_TOK * HDIM);
    dim3 tb(32, 8);
    transpose_cvt_kernel<<<dim3(IDIM / 32, HDIM / 32, EXP), tb, 0, stream>>>(w_gate, wgt, HDIM, IDIM);
    transpose_cvt_kernel<<<dim3(IDIM / 32, HDIM / 32, EXP), tb, 0, stream>>>(w_up,   wut, HDIM, IDIM);
    transpose_cvt_kernel<<<dim3(HDIM / 32, IDIM / 32, EXP), tb, 0, stream>>>(w_down, wdt, IDIM, HDIM);

    // 3. router + aux loss
    router_kernel<<<N_TOK / 8, 256, 0, stream>>>(x, gate_w, cnt, loadv, countsv, etok, ewt);
    aux_kernel<<<1, 32, 0, stream>>>(loadv, countsv, y + (long)N_TOK * HDIM);

    // 4. routed SwiGLU FFN
    moe_ffn_kernel<<<EXP * (N_TOK / 16), 256, 0, stream>>>(xb, wgt, wut, wdt, cnt, etok, ewt, y);
}
